// LSTMCellNM_61924838474036
// MI455X (gfx1250) — compile-verified, matrix operand order checked by reading, not run on hardware
//
#include <hip/hip_runtime.h>
#include <math.h>

// ---------------------------------------------------------------------------
// LSTM cell fused kernel for gfx1250 (MI455X):
//   z = X@Wk + H@Wr + bias ; gates -> (h, c)
// GEMM path: f32 -> bf16 on the fly, v_wmma_f32_16x16x32_bf16, f32 accum.
// Tile: 128 batch rows x 32 units x 4 gates per workgroup (256 thr = 8 waves).
// Double-buffered LDS, coalesced b128 staging, preloaded B fragments.
// ---------------------------------------------------------------------------

typedef __attribute__((ext_vector_type(16))) __bf16 v16bf;
typedef __attribute__((ext_vector_type(8)))  float  v8f;
typedef __attribute__((ext_vector_type(4)))  unsigned int u32x4;
typedef __attribute__((ext_vector_type(4)))  float  f32x4;

#define BATCH 8192
#define DIM   1024
#define UNITS 1024
#define ND    4096         // 4*UNITS
#define BM    128          // batch rows per workgroup
#define BU    32           // units per workgroup (x4 gates -> 128 LDS cols)
#define KT    32           // K per step (one bf16 WMMA)
#define LDSTRIDE 20        // dwords per row/col in LDS (16 data + 4 pad, 16B aligned)
#define NSTEPS 64          // 2 phases x (1024/32)

union FragU {
    u32x4 q[2];
    v16bf v;
};

__device__ __forceinline__ unsigned int pack_bf16(float x, float y) {
    unsigned int ux = __float_as_uint(x), uy = __float_as_uint(y);
    unsigned int rx = (ux + 0x7FFFu + ((ux >> 16) & 1u)) >> 16;   // RNE
    unsigned int ry = (uy + 0x7FFFu + ((uy >> 16) & 1u)) >> 16;
    return rx | (ry << 16);
}

__device__ __forceinline__ float hsig(float x) {
    return fminf(fmaxf(0.2f * x + 0.5f, 0.0f), 1.0f);
}

__global__ __launch_bounds__(256)
void lstm_cell_wmma(const float* __restrict__ X,     // [BATCH, DIM]
                    const float* __restrict__ Hm,    // [BATCH, UNITS]
                    const float* __restrict__ C,     // [BATCH, UNITS]
                    const float* __restrict__ Wk,    // [DIM, ND]
                    const float* __restrict__ Wr,    // [UNITS, ND]
                    const float* __restrict__ bias,  // [ND]
                    float* __restrict__ outH,        // [BATCH, UNITS]
                    float* __restrict__ outC)        // [BATCH, UNITS]
{
    // double-buffered tiles: A = 128 rows x 32 K (bf16 pairs), B = 128 cols x 32 K
    __shared__ __align__(16) unsigned int ldsA[2][BM * LDSTRIDE];       // 2 x 10240 B
    __shared__ __align__(16) unsigned int ldsB[2][4 * BU * LDSTRIDE];   // 2 x 10240 B

    const int t    = threadIdx.x;
    const int lane = t & 31;
    const int wave = t >> 5;
    const int u0   = blockIdx.x * BU;     // unit tile base
    const int b0   = blockIdx.y * BM;     // batch tile base

    const int khalf = lane >> 4;          // fragment K-half select
    const int l15   = lane & 15;

    // acc[ub][g] : 16x16 f32 tile, rows = b0 + wave*16.., cols = u0 + ub*16..
    v8f acc[2][4];
    const v8f vzero = {0.f, 0.f, 0.f, 0.f, 0.f, 0.f, 0.f, 0.f};
#pragma unroll
    for (int ub = 0; ub < 2; ++ub)
#pragma unroll
        for (int g = 0; g < 4; ++g)
            acc[ub][g] = vzero;

    // ---- tile staging: global f32 -> packed bf16 pairs in LDS ----
    auto stage = [&](int s, unsigned int* __restrict__ dA,
                            unsigned int* __restrict__ dB) {
        const float* __restrict__ Asrc = (s < 32) ? X  : Hm;
        const float* __restrict__ Wsrc = (s < 32) ? Wk : Wr;
        const int kk = (s & 31) * KT;

        // A tile: 128 x 32 f32, 1024 float4 / 256 thr = 4 each, coalesced.
#pragma unroll
        for (int i = 0; i < 4; ++i) {
            const int f   = t + 256 * i;
            const int row = f >> 3;       // 0..127
            const int qf  = f & 7;        // float4 index along K
            const f32x4 v = *(const f32x4*)(Asrc + (size_t)(b0 + row) * DIM + kk + qf * 4);
            dA[row * LDSTRIDE + qf * 2 + 0] = pack_bf16(v.x, v.y);
            dA[row * LDSTRIDE + qf * 2 + 1] = pack_bf16(v.z, v.w);
        }

        // B tile: 32(K) x 128 cols (4 gates x 32 units). Coalesced row-major
        // float4 loads of two consecutive k rows; pack K-pairs; swizzled
        // column-major LDS stores. 512 tasks / 256 thr = 2 each.
#pragma unroll
        for (int i = 0; i < 2; ++i) {
            const int f    = t + 256 * i;   // 0..511
            const int kp   = f >> 5;        // K-pair index 0..15
            const int col4 = f & 31;        // float4 column group 0..31
            const int g    = col4 >> 3;
            const int uc4  = col4 & 7;
            const float* p = Wsrc + (size_t)(kk + 2 * kp) * ND
                                  + (size_t)g * UNITS + u0 + uc4 * 4;
            const f32x4 v0 = *(const f32x4*)(p);
            const f32x4 v1 = *(const f32x4*)(p + ND);
            const int q4 = kp >> 2, rem = kp & 3;
#pragma unroll
            for (int e = 0; e < 4; ++e) {
                const int col = 4 * col4 + e;
                const int sw  = (col >> 4) & 3;                 // XOR swizzle
                dB[col * LDSTRIDE + ((q4 ^ sw) << 2) + rem] = pack_bf16(v0[e], v1[e]);
            }
        }
    };

    // ---- compute one K-step: 1 A frag + 8 B frags -> 8 WMMAs ----
    auto compute = [&](const unsigned int* __restrict__ sA,
                       const unsigned int* __restrict__ sB) {
        FragU af;
        {
            const unsigned int* arow = sA + (wave * 16 + l15) * LDSTRIDE;
            af.q[0] = *(const u32x4*)(arow + 4 * khalf);
            af.q[1] = *(const u32x4*)(arow + 8 + 4 * khalf);
        }
        FragU bfr[8];
#pragma unroll
        for (int g = 0; g < 4; ++g) {
#pragma unroll
            for (int ub = 0; ub < 2; ++ub) {
                const int col = g * 32 + ub * 16 + l15;
                const int sw  = (col >> 4) & 3;
                const unsigned int* bcol = sB + col * LDSTRIDE;
                bfr[g * 2 + ub].q[0] = *(const u32x4*)(bcol + ((khalf ^ sw) << 2));
                bfr[g * 2 + ub].q[1] = *(const u32x4*)(bcol + (((2 + khalf) ^ sw) << 2));
            }
        }
#pragma unroll
        for (int g = 0; g < 4; ++g) {
#pragma unroll
            for (int ub = 0; ub < 2; ++ub) {
                acc[ub][g] = __builtin_amdgcn_wmma_f32_16x16x32_bf16(
                    /*neg_a=*/false, af.v, /*neg_b=*/false, bfr[g * 2 + ub].v,
                    /*c_mod=*/(short)0, acc[ub][g],
                    /*reuse_a=*/false, /*reuse_b=*/false);
            }
        }
    };

    // ---- software pipeline: stage(s+1) overlaps compute(s) ----
    stage(0, ldsA[0], ldsB[0]);
    for (int s = 0; s < NSTEPS; ++s) {
        __syncthreads();
        if (s + 1 < NSTEPS)
            stage(s + 1, ldsA[(s + 1) & 1], ldsB[(s + 1) & 1]);
        if (s + 2 < NSTEPS) {   // cache prefetch two steps ahead
            const int s2 = s + 2;
            const float* Ap = (s2 < 32) ? X  : Hm;
            const float* Wp = (s2 < 32) ? Wk : Wr;
            const int kk2 = (s2 & 31) * KT;
            __builtin_prefetch(Ap + (size_t)(b0 + (t & 127)) * DIM + kk2 + (t >> 7) * 16, 0, 1);
            __builtin_prefetch(Wp + (size_t)(kk2 + (t >> 3)) * ND
                                  + (size_t)((t >> 1) & 3) * UNITS + u0 + (t & 1) * 16, 0, 1);
        }
        compute(ldsA[s & 1], ldsB[s & 1]);
    }

    // ---- fused LSTM epilogue ----
    // C/D frag layout: element j of v8f -> row M = j + 8*(lane>=16), col N = lane&15
    const int rbase = 8 * khalf;
#pragma unroll
    for (int ub = 0; ub < 2; ++ub) {
        const int u   = u0 + ub * 16 + l15;
        const float bi  = bias[u];
        const float bff = bias[UNITS + u];
        const float bc  = bias[2 * UNITS + u];
        const float bo  = bias[3 * UNITS + u];
#pragma unroll
        for (int j = 0; j < 8; ++j) {
            const int b = b0 + wave * 16 + rbase + j;
            const size_t idx = (size_t)b * UNITS + u;
            const float z0 = acc[ub][0][j] + bi;
            const float z1 = acc[ub][1][j] + bff;
            const float z2 = acc[ub][2][j] + bc;
            const float z3 = acc[ub][3][j] + bo;
            const float ig = hsig(z0);
            const float fg = hsig(z1);
            const float og = hsig(z3);
            const float cn = fg * C[idx] + ig * tanhf(z2);
            const float hn = og * tanhf(cn);
            outH[idx] = hn;
            outC[idx] = cn;
        }
    }
}

extern "C" void kernel_launch(void* const* d_in, const int* in_sizes, int n_in,
                              void* d_out, int out_size, void* d_ws, size_t ws_size,
                              hipStream_t stream) {
    (void)in_sizes; (void)n_in; (void)out_size; (void)d_ws; (void)ws_size;
    const float* X    = (const float*)d_in[0];   // inputs            [8192,1024]
    const float* H    = (const float*)d_in[1];   // h_tm1             [8192,1024]
    const float* C    = (const float*)d_in[2];   // c_tm1             [8192,1024]
    const float* Wk   = (const float*)d_in[3];   // kernel            [1024,4096]
    const float* Wr   = (const float*)d_in[4];   // recurrent_kernel  [1024,4096]
    const float* bias = (const float*)d_in[5];   // bias              [4096]

    float* outH = (float*)d_out;                       // h first
    float* outC = outH + (size_t)BATCH * UNITS;        // then c

    dim3 grid(UNITS / BU, BATCH / BM);   // 32 x 64 workgroups
    dim3 block(256);
    lstm_cell_wmma<<<grid, block, 0, stream>>>(X, H, C, Wk, Wr, bias, outH, outC);
}
